// ASTER_74139725463627
// MI455X (gfx1250) — matrix operand order repaired and checked by reading, not compiled, not run on hardware
//
#include <hip/hip_runtime.h>
#include <hip/hip_bf16.h>

typedef __attribute__((ext_vector_type(16))) __bf16 v16bf;
typedef __attribute__((ext_vector_type(8)))  __bf16 v8bf;
typedef __attribute__((ext_vector_type(8)))  float  v8f;

#define B_   64
#define L_   384
#define C_   512
#define H_   512
#define E_   256
#define T_   127
#define NC_  250
#define KCAT 1280   // context(512) | emb(256) | h(512)
#define G4H  2048   // 4*H

// ---------------------------------------------------------------------------
// bf16 WMMA GEMM, 16x64 strip per wave (4 independent accumulators):
//   D[m][n] = sum_k A[m][k] * W[n][k]  (+bias1[n]+bias2[n])
// A fragment is loaded once per k-step and reused by 4 v_wmma_f32_16x16x32_bf16
// with independent accumulator chains (no WMMA->WMMA RAW stalls).
// EXEC stays all-ones around the WMMAs (tile guard is wave-uniform).
// ---------------------------------------------------------------------------
__global__ __launch_bounds__(256)
void k_gemm_bf16(const __bf16* __restrict__ A, int lda,
                 const __bf16* __restrict__ W, int ldb,
                 float* __restrict__ D, int ldd,
                 int Mtiles, int Ntiles4, int K,
                 const float* __restrict__ bias1,
                 const float* __restrict__ bias2,
                 int nmax)
{
  const int wave = threadIdx.x >> 5;
  const int lane = threadIdx.x & 31;
  const int tile = blockIdx.x * (blockDim.x >> 5) + wave;
  if (tile >= Mtiles * Ntiles4) return;          // wave-uniform
  const int mT = tile / Ntiles4;
  const int nT = tile % Ntiles4;                 // one nT covers 64 columns
  const int lo = lane & 15;
  const int hi = lane >> 4;

  // A fragment: M = lo, K elems k0+hi*16+0..15
  const __bf16* Ab = A + (size_t)(mT * 16 + lo) * lda + hi * 16;
  // B fragments: N = lo (+16*s), K elems k0+hi*16+0..15 (contiguous in W row)
  const __bf16* Wb = W + (size_t)(nT * 64 + lo) * ldb + hi * 16;
  const size_t wstep = (size_t)16 * ldb;

  v8f acc0 = {}, acc1 = {}, acc2 = {}, acc3 = {};
#pragma unroll 2
  for (int k0 = 0; k0 < K; k0 += 32) {
    union { v16bf v; v8bf h[2]; } ua, ub0, ub1, ub2, ub3;
    ua.h[0] = *reinterpret_cast<const v8bf*>(Ab + k0);
    ua.h[1] = *reinterpret_cast<const v8bf*>(Ab + k0 + 8);
    const __bf16* w0 = Wb + k0;
    ub0.h[0] = *reinterpret_cast<const v8bf*>(w0);
    ub0.h[1] = *reinterpret_cast<const v8bf*>(w0 + 8);
    ub1.h[0] = *reinterpret_cast<const v8bf*>(w0 + wstep);
    ub1.h[1] = *reinterpret_cast<const v8bf*>(w0 + wstep + 8);
    ub2.h[0] = *reinterpret_cast<const v8bf*>(w0 + 2 * wstep);
    ub2.h[1] = *reinterpret_cast<const v8bf*>(w0 + 2 * wstep + 8);
    ub3.h[0] = *reinterpret_cast<const v8bf*>(w0 + 3 * wstep);
    ub3.h[1] = *reinterpret_cast<const v8bf*>(w0 + 3 * wstep + 8);
    acc0 = __builtin_amdgcn_wmma_f32_16x16x32_bf16(false, ua.v, false, ub0.v,
                                                   (short)0, acc0, false, false);
    acc1 = __builtin_amdgcn_wmma_f32_16x16x32_bf16(false, ua.v, false, ub1.v,
                                                   (short)0, acc1, false, false);
    acc2 = __builtin_amdgcn_wmma_f32_16x16x32_bf16(false, ua.v, false, ub2.v,
                                                   (short)0, acc2, false, false);
    acc3 = __builtin_amdgcn_wmma_f32_16x16x32_bf16(false, ua.v, false, ub3.v,
                                                   (short)0, acc3, false, false);
  }

  v8f accs[4] = {acc0, acc1, acc2, acc3};
#pragma unroll
  for (int s = 0; s < 4; ++s) {
    const int n = nT * 64 + s * 16 + lo;
    if (n < nmax) {
      float bsum = 0.f;
      if (bias1) bsum += bias1[n];
      if (bias2) bsum += bias2[n];
#pragma unroll
      for (int r = 0; r < 8; ++r) {
        int m = mT * 16 + hi * 8 + r;            // C/D layout: VGPR r -> row r (+8 hi)
        D[(size_t)m * ldd + n] = accs[s][r] + bsum;
      }
    }
  }
}

// ---------------------------------------------------------------------------
// Elementwise / setup kernels
// ---------------------------------------------------------------------------
__global__ void k_f32_to_bf16(const float* __restrict__ in, __bf16* __restrict__ out, int n) {
  int i = blockIdx.x * blockDim.x + threadIdx.x;
  if (i < n) out[i] = (__bf16)in[i];
}

// Wcat[n][0:768] = W_ih[n], Wcat[n][768:1280] = W_hh[n]
__global__ void k_build_wcat(const float* __restrict__ Wih, const float* __restrict__ Whh,
                             __bf16* __restrict__ out) {
  int i = blockIdx.x * blockDim.x + threadIdx.x;   // [0, 2048*1280)
  int k = i % KCAT, r = i / KCAT;
  float v = (k < C_ + E_) ? Wih[(size_t)r * (C_ + E_) + k]
                          : Whh[(size_t)r * H_ + (k - (C_ + E_))];
  out[i] = (__bf16)v;
}

// Wgen padded 250 -> 256 rows (zeros)
__global__ void k_build_wgenp(const float* __restrict__ Wgen, __bf16* __restrict__ out) {
  int i = blockIdx.x * blockDim.x + threadIdx.x;   // [0, 256*512)
  int r = i >> 9;
  out[i] = (__bf16)((r < NC_) ? Wgen[i] : 0.f);
}

__global__ void k_embed(const int* __restrict__ text, const float* __restrict__ table,
                        float* __restrict__ emb_seq) {
  int i = blockIdx.x * blockDim.x + threadIdx.x;   // [0, B*T*E)
  int e = i & (E_ - 1);
  int bt = i >> 8;
  emb_seq[i] = table[(size_t)text[bt] * E_ + e];
}

__global__ void k_init(float* __restrict__ cst, __bf16* __restrict__ hbf,
                       __bf16* __restrict__ xcat) {
  int i = blockIdx.x * blockDim.x + threadIdx.x;   // [0, B*KCAT)
  if (i < B_ * KCAT) xcat[i] = (__bf16)0.f;
  if (i < B_ * H_) { cst[i] = 0.f; hbf[i] = (__bf16)0.f; }
}

// ---------------------------------------------------------------------------
// Attention: logit[b,l] = wscore . tanh(src_feat[b,l,:] + proj[b,:])
// One wave per (b,l) row; src_feat (50MB) is L2-resident across steps.
// ---------------------------------------------------------------------------
__global__ __launch_bounds__(256)
void k_attn_logits(const float* __restrict__ src_feat, const float* __restrict__ proj,
                   const float* __restrict__ wscore, float* __restrict__ logit) {
  const int wave = threadIdx.x >> 5;
  const int lane = threadIdx.x & 31;
  const int row = blockIdx.x * 8 + wave;           // [0, B*L)
  const int b = row / L_;
  const float* sf = src_feat + (size_t)row * H_;
  const float* pr = proj + (size_t)b * H_;
  float s = 0.f;
#pragma unroll 4
  for (int j = lane; j < H_; j += 32)
    s += wscore[j] * tanhf(sf[j] + pr[j]);
  for (int off = 16; off; off >>= 1) s += __shfl_xor(s, off, 32);
  if (lane == 0) logit[row] = s;
}

__global__ __launch_bounds__(32)
void k_softmax(const float* __restrict__ logit, float* __restrict__ alpha) {
  int b = blockIdx.x, lane = threadIdx.x;
  const float* lg = logit + b * L_;
  float mx = -1e30f;
  for (int l = lane; l < L_; l += 32) mx = fmaxf(mx, lg[l]);
  for (int off = 16; off; off >>= 1) mx = fmaxf(mx, __shfl_xor(mx, off, 32));
  float sm = 0.f;
  for (int l = lane; l < L_; l += 32) sm += expf(lg[l] - mx);
  for (int off = 16; off; off >>= 1) sm += __shfl_xor(sm, off, 32);
  float inv = 1.f / sm;
  for (int l = lane; l < L_; l += 32) alpha[b * L_ + l] = expf(lg[l] - mx) * inv;
}

// context[b,c] = sum_l alpha[b,l]*src[b,l,c]; written as bf16 into xcat[:,0:512]
__global__ __launch_bounds__(256)
void k_context(const float* __restrict__ src, const float* __restrict__ alpha,
               __bf16* __restrict__ xcat) {
  int b = blockIdx.x, t = threadIdx.x;
  const float* sb = src + (size_t)b * L_ * C_;
  const float* al = alpha + b * L_;
  float a0 = 0.f, a1 = 0.f;
  for (int l = 0; l < L_; ++l) {
    float w = al[l];
    a0 += w * sb[(size_t)l * C_ + t];
    a1 += w * sb[(size_t)l * C_ + t + 256];
  }
  xcat[b * KCAT + t]       = (__bf16)a0;
  xcat[b * KCAT + t + 256] = (__bf16)a1;
}

__global__ void k_fill_emb(const float* __restrict__ emb_seq, __bf16* __restrict__ xcat, int t) {
  int b = blockIdx.x, e = threadIdx.x;             // 64 x 256
  xcat[b * KCAT + C_ + e] = (__bf16)emb_seq[((size_t)b * T_ + t) * E_ + e];
}

// LSTM cell update; writes new c, and h as bf16 into hbf (h2h GEMM A),
// xcat[:,768:1280] (gate GEMM A) and hid_bf (classifier GEMM A).
__global__ void k_lstm(const float* __restrict__ gates, float* __restrict__ cst,
                       __bf16* __restrict__ hbf, __bf16* __restrict__ xcat,
                       __bf16* __restrict__ hid_bf, int t) {
  int idx = blockIdx.x * blockDim.x + threadIdx.x; // [0, B*H)
  int b = idx >> 9, j = idx & (H_ - 1);
  const float* g = gates + (size_t)b * G4H;
  float i = 1.f / (1.f + expf(-g[j]));
  float f = 1.f / (1.f + expf(-g[H_ + j]));
  float gg = tanhf(g[2 * H_ + j]);
  float o = 1.f / (1.f + expf(-g[3 * H_ + j]));
  float cn = f * cst[idx] + i * gg;
  float h = o * tanhf(cn);
  cst[idx] = cn;
  __bf16 hb = (__bf16)h;
  hbf[idx] = hb;
  xcat[b * KCAT + C_ + E_ + j] = hb;
  hid_bf[((size_t)b * T_ + t) * H_ + j] = hb;
}

// ---------------------------------------------------------------------------
extern "C" void kernel_launch(void* const* d_in, const int* in_sizes, int n_in,
                              void* d_out, int out_size, void* d_ws, size_t ws_size,
                              hipStream_t stream) {
  (void)in_sizes; (void)n_in; (void)out_size; (void)ws_size;
  const float* src    = (const float*)d_in[0];
  const int*   text   = (const int*)d_in[1];
  const float* emb_tb = (const float*)d_in[2];
  const float* Wi2h   = (const float*)d_in[3];
  const float* Wh2h   = (const float*)d_in[4];
  const float* bh2h   = (const float*)d_in[5];
  const float* wscore = (const float*)d_in[6];
  const float* W_ih   = (const float*)d_in[7];
  const float* b_ih   = (const float*)d_in[8];
  const float* W_hh   = (const float*)d_in[9];
  const float* b_hh   = (const float*)d_in[10];
  const float* Wgen   = (const float*)d_in[11];
  const float* bgen   = (const float*)d_in[12];
  float* out = (float*)d_out;

  char* ws = (char*)d_ws;
  size_t off = 0;
  auto alloc = [&](size_t bytes) -> void* {
    void* p = ws + off;
    off = (off + bytes + 255) & ~(size_t)255;
    return p;
  };
  __bf16* src_bf   = (__bf16*)alloc((size_t)B_ * L_ * C_ * 2);
  float*  src_feat = (float*) alloc((size_t)B_ * L_ * H_ * 4);
  __bf16* Wi2h_bf  = (__bf16*)alloc((size_t)H_ * C_ * 2);
  __bf16* Wh2h_bf  = (__bf16*)alloc((size_t)H_ * H_ * 2);
  __bf16* Wcat_bf  = (__bf16*)alloc((size_t)G4H * KCAT * 2);
  __bf16* Wgenp    = (__bf16*)alloc((size_t)256 * H_ * 2);
  float*  emb_seq  = (float*) alloc((size_t)B_ * T_ * E_ * 4);
  __bf16* hid_bf   = (__bf16*)alloc((size_t)B_ * T_ * H_ * 2);
  __bf16* hbf      = (__bf16*)alloc((size_t)B_ * H_ * 2);
  float*  cst      = (float*) alloc((size_t)B_ * H_ * 4);
  float*  proj     = (float*) alloc((size_t)B_ * H_ * 4);
  float*  logit    = (float*) alloc((size_t)B_ * L_ * 4);
  float*  alpha    = (float*) alloc((size_t)B_ * L_ * 4);
  __bf16* xcat     = (__bf16*)alloc((size_t)B_ * KCAT * 2);
  float*  gates    = (float*) alloc((size_t)B_ * G4H * 4);

  // ---- one-time setup ----
  k_f32_to_bf16<<<(B_ * L_ * C_) / 256, 256, 0, stream>>>(src, src_bf, B_ * L_ * C_);
  k_f32_to_bf16<<<(H_ * C_) / 256, 256, 0, stream>>>(Wi2h, Wi2h_bf, H_ * C_);
  k_f32_to_bf16<<<(H_ * H_) / 256, 256, 0, stream>>>(Wh2h, Wh2h_bf, H_ * H_);
  k_build_wcat<<<(G4H * KCAT) / 256, 256, 0, stream>>>(W_ih, W_hh, Wcat_bf);
  k_build_wgenp<<<(256 * H_) / 256, 256, 0, stream>>>(Wgen, Wgenp);
  k_embed<<<(B_ * T_ * E_) / 256, 256, 0, stream>>>(text, emb_tb, emb_seq);
  k_init<<<(B_ * KCAT + 255) / 256, 256, 0, stream>>>(cst, hbf, xcat);

  // src_feat = src @ Wi2h^T  (M=24576, N=512, K=512): 1536 x 8 strip-tiles
  k_gemm_bf16<<<(1536 * 8) / 8, 256, 0, stream>>>(src_bf, C_, Wi2h_bf, C_,
      src_feat, H_, 1536, 8, C_, nullptr, nullptr, H_);

  // ---- teacher-forced recurrence ----
  for (int t = 0; t < T_; ++t) {
    // proj = h @ Wh2h^T + bh2h  (M=64, N=512, K=512): 4 x 8 strip-tiles
    k_gemm_bf16<<<(4 * 8) / 8, 256, 0, stream>>>(hbf, H_, Wh2h_bf, H_,
        proj, H_, 4, 8, H_, bh2h, nullptr, H_);
    k_attn_logits<<<(B_ * L_) / 8, 256, 0, stream>>>(src_feat, proj, wscore, logit);
    k_softmax<<<B_, 32, 0, stream>>>(logit, alpha);
    k_context<<<B_, 256, 0, stream>>>(src, alpha, xcat);
    k_fill_emb<<<B_, E_, 0, stream>>>(emb_seq, xcat, t);
    // gates = [ctx|emb|h] @ [W_ih|W_hh]^T + b_ih + b_hh  (M=64, N=2048, K=1280)
    k_gemm_bf16<<<(4 * 32) / 8, 256, 0, stream>>>(xcat, KCAT, Wcat_bf, KCAT,
        gates, G4H, 4, 32, KCAT, b_ih, b_hh, G4H);
    k_lstm<<<(B_ * H_) / 256, 256, 0, stream>>>(gates, cst, hbf, xcat, hid_bf, t);
  }

  // probs = hiddens @ Wgen^T + bgen  (M=8128, N=256 padded, store n<250)
  k_gemm_bf16<<<(508 * 4) / 8, 256, 0, stream>>>(hid_bf, H_, Wgenp, H_,
      out, NC_, 508, 4, H_, bgen, nullptr, NC_);
}
